// VQRouter_52347061403862
// MI455X (gfx1250) — compile-verified, matrix operand order checked by reading, not run on hardware
//
#include <hip/hip_runtime.h>
#include <hip/hip_bf16.h>

// ---------------------------------------------------------------------------
// VQ router for MI455X (gfx1250): distances via bf16 WMMA, argmin, gather, loss
//   N = 4096 tokens, D = 768, K = 65536 codes
// ---------------------------------------------------------------------------

typedef __attribute__((ext_vector_type(16))) __bf16 v16bf;
typedef __attribute__((ext_vector_type(8)))  __bf16 v8bf;
typedef __attribute__((ext_vector_type(8)))  float  v8f;

#define N_TOK   4096
#define DIM     768
#define K_CODES 65536
#define KCHUNKS 24            // 768 / 32
#define WAVES   8
#define CODES_PER_WAVE (K_CODES / WAVES)   // 8192
#define TILES_PER_WAVE (CODES_PER_WAVE / 16) // 512

union FragU { v16bf v; v8bf h[2]; };

__device__ __forceinline__ unsigned short f2bf(float x) {
    unsigned int u = __float_as_uint(x);
    unsigned int r = u + 0x7FFFu + ((u >> 16) & 1u);   // round-to-nearest-even
    return (unsigned short)(r >> 16);
}

// --- 1. codebook f32 -> bf16, plus 0.5*||e||^2 per code ---------------------
__global__ void vq_prep_codebook(const float* __restrict__ cb,
                                 unsigned short* __restrict__ cbh,
                                 float* __restrict__ half_esq) {
    int code = blockIdx.x;
    int t = threadIdx.x;
    const float* row = cb + (size_t)code * DIM;
    unsigned short* orow = cbh + (size_t)code * DIM;
    float s = 0.f;
    for (int j = t; j < DIM; j += 256) {
        float v = row[j];
        s += v * v;
        orow[j] = f2bf(v);
    }
    __shared__ float red[256];
    red[t] = s;
    __syncthreads();
    for (int off = 128; off > 0; off >>= 1) {
        if (t < off) red[t] += red[t + off];
        __syncthreads();
    }
    if (t == 0) half_esq[code] = 0.5f * red[0];
}

// --- 2. embeddings f32 -> bf16 ---------------------------------------------
__global__ void vq_prep_z(const float* __restrict__ z,
                          unsigned short* __restrict__ zh, int n) {
    int i = blockIdx.x * blockDim.x + threadIdx.x;
    int stride = gridDim.x * blockDim.x;
    for (; i < n; i += stride) zh[i] = f2bf(z[i]);
}

// --- 3. main: WMMA distance sweep + argmin ---------------------------------
// grid = 256 (one 16-row M tile), block = 256 threads (8 waves).
// wave w sweeps codes [w*8192, (w+1)*8192).
__global__ void __launch_bounds__(256)
vq_argmin_wmma(const unsigned short* __restrict__ zb_u16,
               const unsigned short* __restrict__ cb_u16,
               const float* __restrict__ half_esq,
               int* __restrict__ codes_i32,
               float* __restrict__ codes_f32) {
    const int m0   = blockIdx.x * 16;
    const int lane = threadIdx.x & 31;
    const int wave = threadIdx.x >> 5;
    const int col  = lane & 15;                 // column-in-tile owned by lane
    const int koff = (lane >= 16) ? 8 : 0;      // ISA 16-bit A/B lane K layout

    const __bf16* zb = (const __bf16*)zb_u16;
    const __bf16* cb = (const __bf16*)cb_u16;

    // Preload all 24 A fragments (16x32 bf16 each) into VGPRs.
    // Lane l (<16): row m=l, K = {c*32+0..7} and {c*32+16..23}
    // Lane l (>=16): row m=l-16, K = {c*32+8..15} and {c*32+24..31}
    const __bf16* zrow = zb + (size_t)(m0 + col) * DIM;
    FragU afr[KCHUNKS];
#pragma unroll
    for (int c = 0; c < KCHUNKS; ++c) {
        afr[c].h[0] = *(const v8bf*)(zrow + c * 32 + koff);
        afr[c].h[1] = *(const v8bf*)(zrow + c * 32 + koff + 16);
    }

    float bestv[8];
    int   besti[8];
#pragma unroll
    for (int v = 0; v < 8; ++v) { bestv[v] = -3.4e38f; besti[v] = 0; }

    const int waveColStart = wave * CODES_PER_WAVE;

    for (int t = 0; t < TILES_PER_WAVE; ++t) {
        const int colBase = waveColStart + (t << 4);
        const int code    = colBase + col;
        const __bf16* crow = cb + (size_t)code * DIM;

        // prefetch this lane's codebook row for the NEXT tile (16 codes ahead)
        __builtin_prefetch((const void*)(crow + 16 * DIM), 0, 0);

        v8f acc = {};
#pragma unroll
        for (int c = 0; c < KCHUNKS; ++c) {
            FragU b;
            b.h[0] = *(const v8bf*)(crow + c * 32 + koff);
            b.h[1] = *(const v8bf*)(crow + c * 32 + koff + 16);
            acc = __builtin_amdgcn_wmma_f32_16x16x32_bf16(
                false, afr[c].v, false, b.v, (short)0, acc, false, false);
        }

        // score = z.e - 0.5*||e||^2  (maximize == minimize distance)
        const float e = half_esq[code];
#pragma unroll
        for (int v = 0; v < 8; ++v) {
            float s = acc[v] - e;
            if (s > bestv[v]) { bestv[v] = s; besti[v] = code; }
        }
    }

    // Cross-lane reduce within each 16-lane half (rows 0..7 in lanes 0-15,
    // rows 8..15 in lanes 16-31). Prefer lower code on ties (argmin semantics).
#pragma unroll
    for (int v = 0; v < 8; ++v) {
        float bv = bestv[v];
        int   bi = besti[v];
#pragma unroll
        for (int mask = 8; mask >= 1; mask >>= 1) {
            float ov = __shfl_xor(bv, mask, 16);
            int   oi = __shfl_xor(bi, mask, 16);
            if (ov > bv || (ov == bv && oi < bi)) { bv = ov; bi = oi; }
        }
        bestv[v] = bv;
        besti[v] = bi;
    }

    __shared__ float sval[WAVES][16];
    __shared__ int   sidx[WAVES][16];
    if (col == 0) {
        const int half = lane >> 4;      // 0 or 1
#pragma unroll
        for (int v = 0; v < 8; ++v) {
            const int row = half * 8 + v;
            sval[wave][row] = bestv[v];
            sidx[wave][row] = besti[v];
        }
    }
    __syncthreads();

    if (threadIdx.x < 16) {
        const int row = threadIdx.x;
        float bv = sval[0][row];
        int   bi = sidx[0][row];
        for (int w = 1; w < WAVES; ++w) {
            float ov = sval[w][row];
            int   oi = sidx[w][row];
            if (ov > bv || (ov == bv && oi < bi)) { bv = ov; bi = oi; }
        }
        const int gr = m0 + row;
        codes_i32[gr] = bi;
        codes_f32[gr] = (float)bi;
    }
}

// --- 4. gather quantized (STE forward == codebook[code]) + loss partials ---
__global__ void vq_gather_loss(const float* __restrict__ emb,
                               const float* __restrict__ cb,
                               const int* __restrict__ codes,
                               float* __restrict__ outq,
                               float* __restrict__ acc) {
    const int tok = blockIdx.x;
    const int t = threadIdx.x;
    const int code = codes[tok];
    const float* q = cb + (size_t)code * DIM;
    const float* z = emb + (size_t)tok * DIM;
    float* o = outq + (size_t)tok * DIM;
    float s = 0.f;
    for (int j = t; j < DIM; j += 256) {
        float qq = q[j], zz = z[j];
        o[j] = qq;
        float d = zz - qq;
        s += d * d;
    }
    __shared__ float red[256];
    red[t] = s;
    __syncthreads();
    for (int off = 128; off > 0; off >>= 1) {
        if (t < off) red[t] += red[t + off];
        __syncthreads();
    }
    if (t == 0) atomicAdd(acc, red[0]);
}

// --- 5. finalize loss: (1 + 0.25) * mean((q-z)^2) --------------------------
__global__ void vq_finalize(const float* __restrict__ acc,
                            float* __restrict__ loss_out) {
    if (threadIdx.x == 0 && blockIdx.x == 0)
        loss_out[0] = 1.25f * acc[0] / (float)(N_TOK * DIM);
}

// ---------------------------------------------------------------------------
extern "C" void kernel_launch(void* const* d_in, const int* in_sizes, int n_in,
                              void* d_out, int out_size, void* d_ws, size_t ws_size,
                              hipStream_t stream) {
    (void)in_sizes; (void)n_in; (void)out_size; (void)ws_size;

    const float* emb = (const float*)d_in[0];   // (2,2048,768) f32
    const float* cb  = (const float*)d_in[1];   // (65536,768)  f32

    // workspace layout (bytes)
    char* ws = (char*)d_ws;
    unsigned short* cb_bf16 = (unsigned short*)(ws);                      // 100,663,296
    float*          hesq    = (float*)(ws + 100663296u);                  //     262,144
    unsigned short* z_bf16  = (unsigned short*)(ws + 100925440u);         //   6,291,456
    int*            codes_i = (int*)(ws + 107216896u);                    //      16,384
    float*          acc     = (float*)(ws + 107233280u);                  //           4

    float* out       = (float*)d_out;
    float* out_codes = out;                    // 4096 codes (as f32 values)
    float* out_q     = out + N_TOK;            // 4096*768 quantized (STE fwd)
    float* out_loss  = out + N_TOK + (size_t)N_TOK * DIM;  // scalar

    hipMemsetAsync(acc, 0, sizeof(float), stream);

    vq_prep_codebook<<<K_CODES, 256, 0, stream>>>(cb, cb_bf16, hesq);
    vq_prep_z<<<2048, 256, 0, stream>>>(emb, z_bf16, N_TOK * DIM);

    vq_argmin_wmma<<<N_TOK / 16, 256, 0, stream>>>(z_bf16, cb_bf16, hesq,
                                                   codes_i, out_codes);

    vq_gather_loss<<<N_TOK, 256, 0, stream>>>(emb, cb, codes_i, out_q, acc);
    vq_finalize<<<1, 64, 0, stream>>>(acc, out_loss);
}
